// Network_12970801234422
// MI455X (gfx1250) — 8-bit matrix operand order repaired and checked by reading, compiled, not run on hardware
//
#include <hip/hip_runtime.h>

#define NPROP 3000
#define NCLS  20
#define NP    3072
#define WPB   96
#define NTILES 192
#define KBLK  24
#define KMAX  20
#define TOPK  5
#define MINCOUNT 5
#define KMITERS 25
#define NEGV  (-1e30f)
#define IOUT  0.4f
#define NGT   (NCLS*TOPK)

typedef __attribute__((ext_vector_type(16))) int   v16i;
typedef __attribute__((ext_vector_type(8)))  float v8f;

__global__ void build_graph(const float* __restrict__ boxes,
                            unsigned int* __restrict__ gbits) {
  int idx = blockIdx.x * blockDim.x + threadIdx.x;
  if (idx >= NP * WPB) return;
  int row = idx / WPB;
  int w   = idx % WPB;
  unsigned int word = 0u;
  if (row < NPROP) {
    float ax1 = boxes[row*4+0], ay1 = boxes[row*4+1];
    float ax2 = boxes[row*4+2], ay2 = boxes[row*4+3];
    float areaA = (ax2-ax1+1.f) * (ay2-ay1+1.f);
    int cbase = w * 32;
    for (int b = 0; b < 32; ++b) {
      int col = cbase + b;
      if (col < NPROP) {
        float bx1 = boxes[col*4+0], by1 = boxes[col*4+1];
        float bx2 = boxes[col*4+2], by2 = boxes[col*4+3];
        float areaB = (bx2-bx1+1.f) * (by2-by1+1.f);
        float iw = fmaxf(fminf(ax2,bx2) - fmaxf(ax1,bx1) + 1.f, 0.f);
        float ih = fmaxf(fminf(ay2,by2) - fmaxf(ay1,by1) + 1.f, 0.f);
        float inter = iw * ih;
        float iou = inter / (areaA + areaB - inter);
        if (iou > IOUT) word |= (1u << b);
      }
    }
  }
  gbits[idx] = word;
}

#define SWZF(v, imm) __int_as_float(__builtin_amdgcn_ds_swizzle(__float_as_int(v), (imm)))
#define SWZI(v, imm) __builtin_amdgcn_ds_swizzle((v), (imm))

__device__ __forceinline__ float wsum(float v) {
  v += SWZF(v, 0x041F); v += SWZF(v, 0x081F); v += SWZF(v, 0x101F);
  v += SWZF(v, 0x201F); v += SWZF(v, 0x401F); return v;
}
__device__ __forceinline__ float wmaxf(float v) {
  v = fmaxf(v, SWZF(v, 0x041F)); v = fmaxf(v, SWZF(v, 0x081F)); v = fmaxf(v, SWZF(v, 0x101F));
  v = fmaxf(v, SWZF(v, 0x201F)); v = fmaxf(v, SWZF(v, 0x401F)); return v;
}
__device__ __forceinline__ float wminf(float v) {
  v = fminf(v, SWZF(v, 0x041F)); v = fminf(v, SWZF(v, 0x081F)); v = fminf(v, SWZF(v, 0x101F));
  v = fminf(v, SWZF(v, 0x201F)); v = fminf(v, SWZF(v, 0x401F)); return v;
}
#define WARGMAX_STEP(imm) { float v2 = SWZF(v, imm); int i2 = SWZI(i, imm); \
  if (v2 > v || (v2 == v && i2 < i)) { v = v2; i = i2; } }
__device__ __forceinline__ void wargmax(float& v, int& i) {
  WARGMAX_STEP(0x041F) WARGMAX_STEP(0x081F) WARGMAX_STEP(0x101F)
  WARGMAX_STEP(0x201F) WARGMAX_STEP(0x401F)
}

__device__ __forceinline__ float bsum(float* red, int tid, float v) {
  int lane = tid & 31, wv = tid >> 5;
  v = wsum(v);
  if (lane == 0) red[wv] = v;
  __syncthreads();
  if (wv == 0) { float t = red[lane]; t = wsum(t); if (lane == 0) red[32] = t; }
  __syncthreads();
  float r = red[32]; __syncthreads(); return r;
}
__device__ __forceinline__ float bmax(float* red, int tid, float v) {
  int lane = tid & 31, wv = tid >> 5;
  v = wmaxf(v);
  if (lane == 0) red[wv] = v;
  __syncthreads();
  if (wv == 0) { float t = red[lane]; t = wmaxf(t); if (lane == 0) red[32] = t; }
  __syncthreads();
  float r = red[32]; __syncthreads(); return r;
}
__device__ __forceinline__ float bmin(float* red, int tid, float v) {
  int lane = tid & 31, wv = tid >> 5;
  v = wminf(v);
  if (lane == 0) red[wv] = v;
  __syncthreads();
  if (wv == 0) { float t = red[lane]; t = wminf(t); if (lane == 0) red[32] = t; }
  __syncthreads();
  float r = red[32]; __syncthreads(); return r;
}
__device__ __forceinline__ void bargmax(float* red, int* redi, int tid,
                                        float v, int i, float* ov, int* oi) {
  int lane = tid & 31, wv = tid >> 5;
  wargmax(v, i);
  if (lane == 0) { red[wv] = v; redi[wv] = i; }
  __syncthreads();
  if (wv == 0) {
    float t = red[lane]; int ti = redi[lane];
    wargmax(t, ti);
    if (lane == 0) { red[32] = t; redi[32] = ti; }
  }
  __syncthreads();
  *ov = red[32]; *oi = redi[32]; __syncthreads();
}
__device__ __forceinline__ void bsum6(float* red, int tid, float* a) {
  int lane = tid & 31, wv = tid >> 5;
#pragma unroll
  for (int j = 0; j < 6; ++j) a[j] = wsum(a[j]);
  if (lane == 0) {
#pragma unroll
    for (int j = 0; j < 6; ++j) red[j * 32 + wv] = a[j];
  }
  __syncthreads();
  if (wv == 0) {
#pragma unroll
    for (int j = 0; j < 6; ++j) {
      float t = red[j * 32 + lane]; t = wsum(t);
      if (lane == 0) red[192 + j] = t;
    }
  }
  __syncthreads();
#pragma unroll
  for (int j = 0; j < 6; ++j) a[j] = red[192 + j];
  __syncthreads();
}

__device__ __forceinline__ unsigned int nib2fp8(unsigned int nib) {
  return ((nib * 0x00204081u) & 0x01010101u) * 0x38u;
}

__global__ __launch_bounds__(1024, 1) void cluster_seq(
    const float* __restrict__ boxes,
    const float* __restrict__ cls_prob,
    const float* __restrict__ im_labels,
    const unsigned int* __restrict__ gbits,
    float* __restrict__ gtb, int* __restrict__ vmask) {
  __shared__ float sp[NP];
  __shared__ float sdeg[NP];
  __shared__ float red[256];
  __shared__ int   redi[64];
  __shared__ unsigned int act[WPB];
  __shared__ unsigned int aliveb[WPB];
  __shared__ unsigned int indw[WPB];
  __shared__ __align__(16) unsigned int sB[KBLK * 32];
  __shared__ int sbnz[KBLK];
  __shared__ unsigned char selb[NP];
  __shared__ float sc[3];
  __shared__ float kscore[KMAX];
  __shared__ int   kidx[KMAX];
  __shared__ int   s_count, s_done;

  const int tid  = threadIdx.x;
  const int lane = tid & 31;
  const int wave = tid >> 5;
  const int hsel = (lane & 16) ? 1 : 0;

  if (tid < WPB) aliveb[tid] = (tid < 93) ? 0xFFFFFFFFu
                             : (tid == 93) ? 0x00FFFFFFu : 0u;
  __syncthreads();

  for (int cls = 0; cls < NCLS; ++cls) {
    float lab = im_labels[cls];
    for (int k = 0; k < 3; ++k) {
      int i = tid + k * 1024;
      sp[i] = (i < NPROP) ? fminf(fmaxf(cls_prob[i*NCLS + cls], 1e-9f), 1.0f - 1e-9f) : 0.0f;
    }
    __syncthreads();

    if (lab > 0.5f) {
      float llo = INFINITY, lhi = -INFINITY, ls = 0.f, lc = 0.f;
      for (int k = 0; k < 3; ++k) {
        int i = tid + k * 1024;
        if ((aliveb[i>>5] >> (i&31)) & 1) {
          float v = sp[i];
          llo = fminf(llo, v); lhi = fmaxf(lhi, v); ls += v; lc += 1.f;
        }
      }
      float lo = bmin(red, tid, llo);
      float hi = bmax(red, tid, lhi);
      float ssum = bsum(red, tid, ls);
      float scnt = bsum(red, tid, lc);
      if (tid == 0) { sc[0] = lo; sc[1] = ssum / fmaxf(scnt, 1.f); sc[2] = hi; }
      __syncthreads();

      for (int it = 0; it < KMITERS; ++it) {
        float c0 = sc[0], c1 = sc[1], c2 = sc[2];
        float acc6[6] = {0.f, 0.f, 0.f, 0.f, 0.f, 0.f};
        for (int k = 0; k < 3; ++k) {
          int i = tid + k * 1024;
          if ((aliveb[i>>5] >> (i&31)) & 1) {
            float v = sp[i];
            float d0 = fabsf(v-c0), d1 = fabsf(v-c1), d2 = fabsf(v-c2);
            int a = 0; float bd = d0;
            if (d1 < bd) { bd = d1; a = 1; }
            if (d2 < bd) { a = 2; }
            acc6[a] += v; acc6[3 + a] += 1.f;
          }
        }
        bsum6(red, tid, acc6);
        if (tid == 0) {
          if (acc6[3] > 0.f) sc[0] = acc6[0] / fmaxf(acc6[3], 1.f);
          if (acc6[4] > 0.f) sc[1] = acc6[1] / fmaxf(acc6[4], 1.f);
          if (acc6[5] > 0.f) sc[2] = acc6[2] / fmaxf(acc6[5], 1.f);
        }
        __syncthreads();
      }

      float c0 = sc[0], c1 = sc[1], c2 = sc[2];
      int bestc = 0; { float bv = c0; if (c1 > bv) { bv = c1; bestc = 1; } if (c2 > bv) { bestc = 2; } }
      float anyl = 0.f;
      for (int k = 0; k < 3; ++k) {
        int i = tid + k * 1024;
        bool al = (aliveb[i>>5] >> (i&31)) & 1;
        float v = sp[i];
        float d0 = fabsf(v-c0), d1 = fabsf(v-c1), d2 = fabsf(v-c2);
        int a = 0; float bd = d0;
        if (d1 < bd) { bd = d1; a = 1; }
        if (d2 < bd) { a = 2; }
        unsigned char s = (al && a == bestc) ? 1 : 0;
        selb[i] = s; anyl += (float)s;
      }
      float anySel = bsum(red, tid, anyl);
      float fv; int fi;
      {
        float lv = -3.0e38f; int li = 0x7FFFFFFF;
        for (int k = 0; k < 3; ++k) {
          int i = tid + k * 1024;
          bool al = (aliveb[i>>5] >> (i&31)) & 1;
          float v = al ? sp[i] : NEGV;
          if (v > lv || (v == lv && i < li)) { lv = v; li = i; }
        }
        bargmax(red, redi, tid, lv, li, &fv, &fi);
      }
      if (anySel < 0.5f) {
        for (int k = 0; k < 3; ++k) selb[tid + k * 1024] = 0;
        __syncthreads();
        if (tid == 0 && ((aliveb[fi>>5] >> (fi&31)) & 1)) selb[fi] = 1;
        __syncthreads();
      }
      if (tid < WPB) {
        unsigned int wrd = 0;
        for (int b = 0; b < 32; ++b) wrd |= ((unsigned int)selb[tid*32 + b]) << b;
        act[tid] = wrd;
      }
      __syncthreads();
      float cntf = bsum(red, tid, (tid < WPB) ? (float)__popc(act[tid]) : 0.f);
      if (tid == 0) { s_count = (int)cntf; s_done = 0; }
      if (tid < KMAX) { kscore[tid] = NEGV; kidx[tid] = 0; }
      __syncthreads();

      for (int step = 0; step < KMAX; ++step) {
        if (tid < KBLK * 32) {
          int kb = tid >> 5, r = tid & 31, hs = r >> 4, v = r & 15;
          int K = (v >> 2) * 32 + ((v >> 1) & 1) * 16 + hs * 8 + (v & 1) * 4;
          unsigned int w = act[kb * 4 + (K >> 5)];
          sB[tid] = nib2fp8((w >> (K & 31)) & 0xFu);
        }
        if (tid < KBLK)
          sbnz[tid] = (act[tid*4] | act[tid*4+1] | act[tid*4+2] | act[tid*4+3]) ? 1 : 0;
        __syncthreads();

        for (int tile = wave; tile < NTILES; tile += 32) {
          unsigned int rmask = (act[tile >> 1] >> ((tile & 1) * 16)) & 0xFFFFu;
          if (rmask == 0u) continue;
          v8f acc = {};
          int row = tile * 16 + (lane & 15);
          const uint4* g4 = (const uint4*)(gbits + (size_t)row * WPB);
          const int hoff = hsel * 8;
          for (int kb = 0; kb < KBLK; ++kb) {
            if (!sbnz[kb]) continue;
            uint4 aw = g4[kb];
            v16i A, B;
#pragma unroll
            for (int v = 0; v < 16; ++v) {
              int klo = ((v >= 8) ? 64 : 0) + (((v & 7) >> 1) * 16) + ((v & 1) * 4);
              unsigned int w = (klo < 32) ? aw.x : (klo < 64) ? aw.y : (klo < 96) ? aw.z : aw.w;
              A[v] = (int)nib2fp8((w >> ((klo & 31) + hoff)) & 0xFu);
            }
            const uint4* b4 = (const uint4*)(sB + kb * 32 + hsel * 16);
            uint4 b0 = b4[0], b1 = b4[1], b2 = b4[2], b3 = b4[3];
            B[0] = (int)b0.x; B[1] = (int)b0.y; B[2]  = (int)b0.z; B[3]  = (int)b0.w;
            B[4] = (int)b1.x; B[5] = (int)b1.y; B[6]  = (int)b1.z; B[7]  = (int)b1.w;
            B[8] = (int)b2.x; B[9] = (int)b2.y; B[10] = (int)b2.z; B[11] = (int)b2.w;
            B[12] = (int)b3.x; B[13] = (int)b3.y; B[14] = (int)b3.z; B[15] = (int)b3.w;
            acc = __builtin_amdgcn_wmma_f32_16x16x128_fp8_fp8(
                A, B, (short)0, acc, false, false);
          }
          if (lane == 0) {
#pragma unroll
            for (int j = 0; j < 8; ++j) sdeg[tile*16 + j] = acc[j];
          } else if (lane == 16) {
#pragma unroll
            for (int j = 0; j < 8; ++j) sdeg[tile*16 + 8 + j] = acc[j];
          }
        }
        __syncthreads();

        float lv = -3.0e38f; int li = 0x7FFFFFFF;
        for (int k = 0; k < 3; ++k) {
          int i = tid + k * 1024;
          bool ac = (act[i>>5] >> (i&31)) & 1;
          float v = ac ? sdeg[i] : -1.0f;
          if (v > lv || (v == lv && i < li)) { lv = v; li = i; }
        }
        float dv; int tmp;
        bargmax(red, redi, tid, lv, li, &dv, &tmp);

        int rloc = 0;
        if (tid < WPB) {
          unsigned int u = gbits[(size_t)tmp * WPB + tid] & act[tid];
          indw[tid] = u;
          act[tid] &= ~u;
          rloc = __popc(u);
        }
        float removed = bsum(red, tid, (float)rloc);
        float sl = NEGV;
        for (int k = 0; k < 3; ++k) {
          int i = tid + k * 1024;
          if ((indw[i>>5] >> (i&31)) & 1) sl = fmaxf(sl, sp[i]);
        }
        float score = bmax(red, tid, sl);
        if (tid == 0) {
          kidx[step] = tmp; kscore[step] = score;
          s_count -= (int)removed;
          if (s_count <= MINCOUNT) s_done = 1;
        }
        __syncthreads();
        if (s_done) break;
      }

      if (tid == 0) {
        bool used[KMAX];
        for (int j = 0; j < KMAX; ++j) used[j] = false;
        for (int r = 0; r < TOPK; ++r) {
          float bv = -3.0e38f; int bj = 0;
          for (int j = 0; j < KMAX; ++j)
            if (!used[j] && kscore[j] > bv) { bv = kscore[j]; bj = j; }
          used[bj] = true;
          int si = kidx[bj];
          int slot = cls * TOPK + r;
          int val = (bv > NEGV * 0.5f) ? 1 : 0;
          vmask[slot] = val;
          gtb[slot*4+0] = boxes[si*4+0];
          gtb[slot*4+1] = boxes[si*4+1];
          gtb[slot*4+2] = boxes[si*4+2];
          gtb[slot*4+3] = boxes[si*4+3];
          if (val) aliveb[si>>5] &= ~(1u << (si & 31));
        }
      }
      __syncthreads();
    } else {
      if (tid == 0) {
        for (int r = 0; r < TOPK; ++r) {
          int slot = cls * TOPK + r;
          vmask[slot] = 0;
          gtb[slot*4+0] = 0.f; gtb[slot*4+1] = 0.f;
          gtb[slot*4+2] = 0.f; gtb[slot*4+3] = 0.f;
        }
      }
      __syncthreads();
    }
  }
}

__global__ void final_overlaps(const float* __restrict__ boxes,
                               const float* __restrict__ gtb,
                               const int* __restrict__ vmask,
                               float* __restrict__ out) {
  __shared__ float sg[NGT*4];
  __shared__ int   sv[NGT];
  int tid = threadIdx.x;
  for (int j = tid; j < NGT; j += blockDim.x) {
    sv[j] = vmask[j];
    sg[j*4+0] = gtb[j*4+0]; sg[j*4+1] = gtb[j*4+1];
    sg[j*4+2] = gtb[j*4+2]; sg[j*4+3] = gtb[j*4+3];
  }
  __syncthreads();
  int i = blockIdx.x * blockDim.x + tid;
  if (i < NPROP) {
    float ax1 = boxes[i*4+0], ay1 = boxes[i*4+1];
    float ax2 = boxes[i*4+2], ay2 = boxes[i*4+3];
    float areaA = (ax2-ax1+1.f) * (ay2-ay1+1.f);
    float m = -1.0f;
    for (int j = 0; j < NGT; ++j) {
      if (sv[j]) {
        float bx1 = sg[j*4+0], by1 = sg[j*4+1];
        float bx2 = sg[j*4+2], by2 = sg[j*4+3];
        float areaB = (bx2-bx1+1.f) * (by2-by1+1.f);
        float iw = fmaxf(fminf(ax2,bx2) - fmaxf(ax1,bx1) + 1.f, 0.f);
        float ih = fmaxf(fminf(ay2,by2) - fmaxf(ay1,by1) + 1.f, 0.f);
        float inter = iw * ih;
        float iou = inter / (areaA + areaB - inter);
        m = fmaxf(m, iou);
      }
    }
    out[i] = m;
  }
}

extern "C" void kernel_launch(void* const* d_in, const int* in_sizes, int n_in,
                              void* d_out, int out_size, void* d_ws, size_t ws_size,
                              hipStream_t stream) {
  const float* boxes     = (const float*)d_in[0];
  const float* cls_prob  = (const float*)d_in[1];
  const float* im_labels = (const float*)d_in[2];

  unsigned int* gbits = (unsigned int*)d_ws;
  float* gtb  = (float*)(gbits + (size_t)NP * WPB);
  int*   vmsk = (int*)(gtb + NGT * 4);

  int totw = NP * WPB;
  build_graph<<<(totw + 255) / 256, 256, 0, stream>>>(boxes, gbits);
  cluster_seq<<<1, 1024, 0, stream>>>(boxes, cls_prob, im_labels, gbits, gtb, vmsk);
  final_overlaps<<<(NPROP + 255) / 256, 256, 0, stream>>>(boxes, gtb, vmsk, (float*)d_out);
}
